// GraphSAGELinkPredictor_70523363000941
// MI455X (gfx1250) — matrix operand order repaired and checked by reading, not compiled, not run on hardware
//
#include <hip/hip_runtime.h>
#include <hip/hip_bf16.h>

typedef __attribute__((ext_vector_type(16))) __bf16 v16bf;
typedef __attribute__((ext_vector_type(8)))  float  v8f;

#define N_USERS   100000
#define N_RECIPES 50000
#define N_EDGES   600000
#define N_LABELS  200000
#define HDIM      128
#define OUTDIM    64

// ---------------------------------------------------------------------------
// A-operand loader from a precomputed per-lane base pointer.
// base = X + gm*ldx + half*16; step offset kb is compile-time after unroll, so
// all 4 b128 loads use immediate offsets.  A layout (matches B):
// elements 0..15 <- K = kb + half*16 + i.
__device__ inline v16bf load_a_ptr(const float* __restrict__ p, int kb, float s) {
    v16bf a;
#pragma unroll
    for (int i = 0; i < 8; ++i)  a[i]     = (__bf16)(p[kb + i] * s);
#pragma unroll
    for (int i = 0; i < 8; ++i)  a[8 + i] = (__bf16)(p[kb + 8 + i] * s);
    return a;
}

// B-operand loader from a precomputed per-lane base pointer.
// base = W + gn*ldw + half*16; elements 0..15 <- K = kb + half*16 + i.
__device__ inline v16bf load_b_ptr(const float* __restrict__ p, int kb) {
    v16bf b;
#pragma unroll
    for (int i = 0; i < 16; ++i) b[i] = (__bf16)p[kb + i];
    return b;
}

__device__ inline v8f wmma_bf16(v16bf a, v16bf b, v8f c) {
    return __builtin_amdgcn_wmma_f32_16x16x32_bf16(
        /*neg_a=*/false, a, /*neg_b=*/false, b,
        /*c_mod=*/(short)0, c, /*reuse_a=*/false, /*reuse_b=*/false);
}

// ---------------------------------------------------------------------------
// Fused GEMM:
//   O[m,n] = act( sum_{k<K1} A1[m,k]*scale[m]*W1[n,k]
//               + sum_{k<K2} A2[m,k]*W2[n,k] + bias[n] )
// scale[m] = 1/max(cnt[m],1) when cnt != nullptr (SAGE scatter-mean).
// One wave32 per block computes a 16x64 output strip: per K-step, one A tile
// and four B tiles are loaded into distinct registers (clause-able, immediate
// offsets), then 4 independent v_wmma_f32_16x16x32_bf16 issue back-to-back.
// K1/K2/NN compile-time -> fully unrolled, no branches, EXEC all-1s (no tails).
template <int K1, int K2, int NN>
__global__ void __launch_bounds__(32)
wmma_gemm_kernel(const float* __restrict__ X1, const float* __restrict__ cnt,
                 const float* __restrict__ X2,
                 const float* __restrict__ W1, const float* __restrict__ W2,
                 const float* __restrict__ bias, float* __restrict__ O,
                 int relu) {
    const int lane = threadIdx.x & 31;
    const int half = lane >> 4;
    const int lidx = lane & 15;

    const int mtile = blockIdx.x;
    const int gm  = mtile * 16 + lidx;      // A row owned by this lane
    const int gn0 = blockIdx.y * 64 + lidx; // first of 4 output cols owned by lane

    float scale = 1.0f;
    if (cnt) scale = 1.0f / fmaxf(cnt[gm], 1.0f);

    // Precomputed per-lane base pointers (all K offsets fold into imm offsets).
    const float* pa1 = X1 + (size_t)gm * K1 + half * 16;
    const float* pb1[4];
#pragma unroll
    for (int j = 0; j < 4; ++j)
        pb1[j] = W1 + (size_t)(gn0 + 16 * j) * K1 + half * 16;

    v8f acc[4] = {{}, {}, {}, {}};

#pragma unroll
    for (int kb = 0; kb < K1; kb += 32) {
        v16bf a  = load_a_ptr(pa1, kb, scale);
        v16bf b0 = load_b_ptr(pb1[0], kb);
        v16bf b1 = load_b_ptr(pb1[1], kb);
        v16bf b2 = load_b_ptr(pb1[2], kb);
        v16bf b3 = load_b_ptr(pb1[3], kb);
        acc[0] = wmma_bf16(a, b0, acc[0]);
        acc[1] = wmma_bf16(a, b1, acc[1]);
        acc[2] = wmma_bf16(a, b2, acc[2]);
        acc[3] = wmma_bf16(a, b3, acc[3]);
    }
    if constexpr (K2 > 0) {
        const float* pa2 = X2 + (size_t)gm * K2 + half * 16;
        const float* pb2[4];
#pragma unroll
        for (int j = 0; j < 4; ++j)
            pb2[j] = W2 + (size_t)(gn0 + 16 * j) * K2 + half * 16;
#pragma unroll
        for (int kb = 0; kb < K2; kb += 32) {
            v16bf a  = load_a_ptr(pa2, kb, 1.0f);
            v16bf b0 = load_b_ptr(pb2[0], kb);
            v16bf b1 = load_b_ptr(pb2[1], kb);
            v16bf b2 = load_b_ptr(pb2[2], kb);
            v16bf b3 = load_b_ptr(pb2[3], kb);
            acc[0] = wmma_bf16(a, b0, acc[0]);
            acc[1] = wmma_bf16(a, b1, acc[1]);
            acc[2] = wmma_bf16(a, b2, acc[2]);
            acc[3] = wmma_bf16(a, b3, acc[3]);
        }
    }

    // Epilogue: D layout row = half*8 + v, col = gn0 + 16j.  Base pointer with
    // compile-time NN -> immediate-offset stores.
    float* po = O + (size_t)(mtile * 16 + half * 8) * NN + gn0;
#pragma unroll
    for (int j = 0; j < 4; ++j) {
        const float bv = bias[gn0 + 16 * j];
#pragma unroll
        for (int v = 0; v < 8; ++v) {
            float o = acc[j][v] + bv;
            if (relu) o = fmaxf(o, 0.0f);
            po[(size_t)v * NN + 16 * j] = o;
        }
    }
}

// ---------------------------------------------------------------------------
// Edge scatter: for each edge e (src user s, dst recipe d):
//   agg_r[d,:] += fuser[s,:]; cnt_r[d] += 1;
//   agg_u[s,:] += frec[d,:];  cnt_u[s] += 1;
// blockDim = (128, 2): x indexes channel, y indexes edge within block.
__global__ void __launch_bounds__(256)
scatter_add_kernel(const float* __restrict__ fuser, const float* __restrict__ frec,
                   const int* __restrict__ esrc, const int* __restrict__ edst,
                   float* __restrict__ agg_r, float* __restrict__ cnt_r,
                   float* __restrict__ agg_u, float* __restrict__ cnt_u) {
    const int e = blockIdx.x * 2 + threadIdx.y;
    const int c = threadIdx.x;
    const int s = esrc[e];
    const int d = edst[e];
    atomicAdd(&agg_r[(size_t)d * HDIM + c], fuser[(size_t)s * HDIM + c]);
    atomicAdd(&agg_u[(size_t)s * HDIM + c], frec[(size_t)d * HDIM + c]);
    if (c == 0) {
        atomicAdd(&cnt_r[d], 1.0f);
        atomicAdd(&cnt_u[s], 1.0f);
    }
}

__global__ void __launch_bounds__(256)
zero_kernel(float* __restrict__ p, long n) {
    long i = (long)blockIdx.x * 256 + threadIdx.x;
    if (i < n) p[i] = 0.0f;
}

// ---------------------------------------------------------------------------
// Decoder: one wave32 per label pair; 2 channels/lane of the 64-dim embeddings,
// butterfly reduction, cosine similarity.
__global__ void __launch_bounds__(256)
decoder_kernel(const float* __restrict__ zu, const float* __restrict__ zr,
               const int* __restrict__ lsrc, const int* __restrict__ ldst,
               float* __restrict__ out) {
    const int lane = threadIdx.x & 31;
    const int wv   = threadIdx.x >> 5;
    const int p    = blockIdx.x * 8 + wv;
    const float* a = zu + (size_t)lsrc[p] * OUTDIM + lane * 2;
    const float* b = zr + (size_t)ldst[p] * OUTDIM + lane * 2;
    float a0 = a[0], a1 = a[1], b0 = b[0], b1 = b[1];
    float sa = a0 * a0 + a1 * a1;
    float sb = b0 * b0 + b1 * b1;
    float dt = a0 * b0 + a1 * b1;
#pragma unroll
    for (int m = 16; m >= 1; m >>= 1) {
        sa += __shfl_xor(sa, m, 32);
        sb += __shfl_xor(sb, m, 32);
        dt += __shfl_xor(dt, m, 32);
    }
    if (lane == 0)
        out[p] = dt / (fmaxf(sqrtf(sa), 1e-12f) * fmaxf(sqrtf(sb), 1e-12f));
}

// ---------------------------------------------------------------------------
extern "C" void kernel_launch(void* const* d_in, const int* in_sizes, int n_in,
                              void* d_out, int out_size, void* d_ws, size_t ws_size,
                              hipStream_t stream) {
    const float* x_user   = (const float*)d_in[0];
    const float* x_recipe = (const float*)d_in[1];
    const int*   esrc     = (const int*)d_in[2];
    const int*   edst     = (const int*)d_in[3];
    const int*   lsrc     = (const int*)d_in[4];
    const int*   ldst     = (const int*)d_in[5];
    const float* Wu     = (const float*)d_in[6];
    const float* bu     = (const float*)d_in[7];
    const float* Wrec   = (const float*)d_in[8];
    const float* brec   = (const float*)d_in[9];
    const float* c1urWl = (const float*)d_in[10];
    const float* c1urbl = (const float*)d_in[11];
    const float* c1urWr = (const float*)d_in[12];
    const float* c1ruWl = (const float*)d_in[13];
    const float* c1rubl = (const float*)d_in[14];
    const float* c1ruWr = (const float*)d_in[15];
    const float* c2urWl = (const float*)d_in[16];
    const float* c2urbl = (const float*)d_in[17];
    const float* c2urWr = (const float*)d_in[18];
    const float* c2ruWl = (const float*)d_in[19];
    const float* c2rubl = (const float*)d_in[20];
    const float* c2ruWr = (const float*)d_in[21];
    float* out = (float*)d_out;

    // Workspace carving (floats). zu/zr reuse hu/hr once those are dead.
    float* ws    = (float*)d_ws;
    float* hu    = ws; ws += (size_t)N_USERS * HDIM;
    float* hr    = ws; ws += (size_t)N_RECIPES * HDIM;
    float* u1    = ws; ws += (size_t)N_USERS * HDIM;
    float* r1    = ws; ws += (size_t)N_RECIPES * HDIM;
    float* agg_u = ws; ws += (size_t)N_USERS * HDIM;
    float* agg_r = ws; ws += (size_t)N_RECIPES * HDIM;
    float* cnt_u = ws; ws += (size_t)N_USERS;
    float* cnt_r = ws; ws += (size_t)N_RECIPES;
    float* zu = hu;
    float* zr = hr;

    const long aggN = (long)N_USERS * HDIM + (long)N_RECIPES * HDIM
                    + N_USERS + N_RECIPES;               // agg_u..cnt_r contiguous
    const dim3 blkW(32), blkS(128, 2), blk256(256);
    const int MU = N_USERS / 16, MR = N_RECIPES / 16;    // exact: 6250, 3125

    // ---- input projections -------------------------------------------------
    wmma_gemm_kernel<128, 0, HDIM><<<dim3(MU, 2), blkW, 0, stream>>>(
        x_user, nullptr, nullptr, Wu, nullptr, bu, hu, 0);
    wmma_gemm_kernel<256, 0, HDIM><<<dim3(MR, 2), blkW, 0, stream>>>(
        x_recipe, nullptr, nullptr, Wrec, nullptr, brec, hr, 0);

    // ---- conv1: scatter-mean + fused SAGE GEMM (ReLU) ----------------------
    zero_kernel<<<(aggN + 255) / 256, blk256, 0, stream>>>(agg_u, aggN);
    scatter_add_kernel<<<N_EDGES / 2, blkS, 0, stream>>>(
        hu, hr, esrc, edst, agg_r, cnt_r, agg_u, cnt_u);
    wmma_gemm_kernel<128, 128, HDIM><<<dim3(MR, 2), blkW, 0, stream>>>(
        agg_r, cnt_r, hr, c1urWl, c1urWr, c1urbl, r1, 1);
    wmma_gemm_kernel<128, 128, HDIM><<<dim3(MU, 2), blkW, 0, stream>>>(
        agg_u, cnt_u, hu, c1ruWl, c1ruWr, c1rubl, u1, 1);

    // ---- conv2 -------------------------------------------------------------
    zero_kernel<<<(aggN + 255) / 256, blk256, 0, stream>>>(agg_u, aggN);
    scatter_add_kernel<<<N_EDGES / 2, blkS, 0, stream>>>(
        u1, r1, esrc, edst, agg_r, cnt_r, agg_u, cnt_u);
    wmma_gemm_kernel<128, 128, OUTDIM><<<dim3(MR, 1), blkW, 0, stream>>>(
        agg_r, cnt_r, r1, c2urWl, c2urWr, c2urbl, zr, 0);
    wmma_gemm_kernel<128, 128, OUTDIM><<<dim3(MU, 1), blkW, 0, stream>>>(
        agg_u, cnt_u, u1, c2ruWl, c2ruWr, c2rubl, zu, 0);

    // ---- decoder -----------------------------------------------------------
    decoder_kernel<<<N_LABELS / 8, blk256, 0, stream>>>(zu, zr, lsrc, ldst, out);
}
